// MPLayer_34273839022285
// MI455X (gfx1250) — matrix operand order repaired and checked by reading, not compiled, not run on hardware
//
#include <hip/hip_runtime.h>

typedef _Float16 v8h  __attribute__((ext_vector_type(8)));
typedef _Float16 v16h __attribute__((ext_vector_type(16)));
typedef float    v8f  __attribute__((ext_vector_type(8)));
typedef unsigned int v4u __attribute__((ext_vector_type(4)));
typedef int      v8i  __attribute__((ext_vector_type(8)));
typedef int      v4i  __attribute__((ext_vector_type(4)));

#define BB 2
#define NN 50000
#define EE 400000
#define HH 128

// Fragment-major weight layout: frag(kt,t) = 512 f16 (1 KB) block.
//   half0: lane L holds K = kt*32 + 16*(L>>4) + {0..7}   at block + L*8
//   half1: lane L holds K = kt*32 + 16*(L>>4) + 8+{0..7} at block + 256 + L*8
// ws weight-area element offsets (f16 units)
#define OFF_WE2T 49152                 // We1t: 96 frags  (K=384)
#define OFF_WN1T 65536                 // We2t: 32 frags  (K=128)
#define OFF_WN2T 98304                 // Wn1t: 64 frags  (K=256); Wn2t: 32 frags

// ws byte offsets
#define WS_XH    0ull
#define WS_AGG   25600000ull          // after B*N*H*2 bytes of xh
#define WS_WT    76800000ull          // after B*N*H*4 bytes of agg

// LDS element offsets (f16 units)
#define LDS_E_SC   49152               // edge: scratch after We1t
#define LDS_N_WN2  32768
#define LDS_N_SC   49152               // node: scratch after Wn1t+Wn2t

__device__ __forceinline__ v8f wmma_f16(v16h a, v16h b, v8f c) {
    return __builtin_amdgcn_wmma_f32_16x16x32_f16(
        false, a, false, b, (short)0, c, false, false);
}

__device__ __forceinline__ v16h joinh(v8h lo, v8h hi) {
    return __builtin_shufflevector(lo, hi, 0,1,2,3,4,5,6,7,8,9,10,11,12,13,14,15);
}

// load fragment t from a frag-major table (LDS or global)
__device__ __forceinline__ v16h ldfrag(const _Float16* fb, int t, int lane) {
    return joinh(*(const v8h*)(fb + t * 512 + lane * 8),
                 *(const v8h*)(fb + t * 512 + 256 + lane * 8));
}

// ---- TDM: 1-D DMA of `bytes` (multiple of 8) from global to LDS offset ----
__device__ __forceinline__ void tdm_load_1d(const void* gsrc, unsigned ldsOff, unsigned bytes) {
    unsigned units = bytes >> 3;                       // data_size = 3 -> 8-byte units
    unsigned long long ga = (unsigned long long)gsrc;
    v4u g0;
    g0[0] = 1u;                                        // count=1 (valid user descriptor)
    g0[1] = ldsOff;                                    // lds_addr (bytes)
    g0[2] = (unsigned)ga;                              // global_addr[31:0]
    g0[3] = (unsigned)((ga >> 32) & 0x1FFFFFFull) | (2u << 30);  // addr[56:32] | type=2
    v8i g1;
    g1[0] = 0x30000;                                   // workgroup_mask=0, data_size=3
    g1[1] = (int)((units & 0xFFFFu) << 16);            // tensor_dim0[15:0] -> bits[63:48]
    g1[2] = (int)(((units >> 16) & 0xFFFFu) | (1u << 16)); // tensor_dim0[31:16], tensor_dim1=1
    g1[3] = (int)((units & 0xFFFFu) << 16);            // tensor_dim1 hi=0, tile_dim0=units
    g1[4] = 0;                                         // tile_dim1=0, tile_dim2=0 (unused)
    g1[5] = (int)units;                                // tensor_dim0_stride[31:0]
    g1[6] = 0;
    g1[7] = 0;
    v4i z4 = {0, 0, 0, 0};
    v8i z8 = {0, 0, 0, 0, 0, 0, 0, 0};
    __builtin_amdgcn_tensor_load_to_lds(g0, g1, z4, z4, z8, 0);
}

// ---------------- weight prep: f32 [K][128] -> f16 frag-major W^T ----------------
__global__ void prep_weights_kernel(const float* __restrict__ We1,
                                    const float* __restrict__ We2,
                                    const float* __restrict__ Wn1,
                                    const float* __restrict__ Wn2,
                                    _Float16* __restrict__ wt) {
    const float* src; _Float16* dst; int K;
    switch (blockIdx.y) {
        case 0:  src = We1; dst = wt;            K = 384; break;
        case 1:  src = We2; dst = wt + OFF_WE2T; K = 128; break;
        case 2:  src = Wn1; dst = wt + OFF_WN1T; K = 256; break;
        default: src = Wn2; dst = wt + OFF_WN2T; K = 128; break;
    }
    int id = blockIdx.x * blockDim.x + threadIdx.x;
    if (id >= 128 * K) return;
    int frag = id >> 9;            // (kt*8 + t)
    int e    = id & 511;
    int half = e >> 8;             // 0 or 1
    int L    = (e >> 3) & 31;      // lane
    int j    = e & 7;
    int t    = frag & 7;
    int kt   = frag >> 3;
    int h = 16 * t + (L & 15);
    int k = kt * 32 + 16 * (L >> 4) + half * 8 + j;
    dst[id] = (_Float16)src[(size_t)k * HH + h];
}

// ---------------- x: f32 -> f16 ----------------
__global__ void prep_x_kernel(const float* __restrict__ x, _Float16* __restrict__ xh) {
    long long i = ((long long)blockIdx.x * blockDim.x + threadIdx.x) * 8;
    if (i >= (long long)BB * NN * HH) return;
    v8f f = *(const v8f*)(x + i);
    v8h h;
#pragma unroll
    for (int j = 0; j < 8; ++j) h[j] = (_Float16)f[j];
    *(v8h*)(xh + i) = h;
}

// ---------------- edge MLP + scatter-add ----------------
__global__ __launch_bounds__(256, 4) void edge_kernel(
    const _Float16* __restrict__ xh,
    const long long* __restrict__ eidx,     // [2][E]
    const float* __restrict__ edge_attr,    // [B][E][H]
    const _Float16* __restrict__ We1t_g,    // frag-major (TDM source)
    const float* __restrict__ be1,
    const _Float16* __restrict__ We2t_g,    // frag-major (global/L2)
    const float* __restrict__ be2,
    float* __restrict__ agg,                // [B][N][H]
    float* __restrict__ eout)               // [B][E][H]
{
    extern __shared__ _Float16 smem[];      // [We1t: 49152][scratch: 8*2304]
    const int wave = threadIdx.x >> 5;
    const int lane = threadIdx.x & 31;
    const int m    = lane & 15;
    const int lh   = lane >> 4;

    // stage layer-1 weights into LDS via the Tensor Data Mover (one wave issues)
    if (wave == 0) {
        tdm_load_1d(We1t_g, (unsigned)(unsigned long long)(void*)smem, 49152 * 2);
        __builtin_amdgcn_s_wait_tensorcnt(0);
    }
    __syncthreads();

    const _Float16* We1s = smem;
    _Float16* sc = smem + LDS_E_SC + wave * (16 * 144);

#pragma unroll 1
    for (int g = 0; g < 5; ++g) {
        const int tile = (g * 1250 + blockIdx.x) * 8 + wave;   // 0..49999 exact
        const int b    = tile / (EE / 16);
        const int e0   = (tile % (EE / 16)) * 16;
        const int eI   = e0 + m;

        const int si = (int)eidx[eI];
        const int di = (int)eidx[EE + eI];
        const _Float16* rowS = xh + ((size_t)b * NN + si) * HH;
        const _Float16* rowD = xh + ((size_t)b * NN + di) * HH;
        const float*    rowA = edge_attr + ((size_t)b * EE + eI) * HH;

        v8f acc[8] = {};
#pragma unroll
        for (int kt = 0; kt < 12; ++kt) {
            const int k0 = kt * 32 + lh * 16;
            v16h bfrag;
            if (kt < 4) {
                bfrag = *(const v16h*)(rowS + k0);
            } else if (kt < 8) {
                bfrag = *(const v16h*)(rowD + (k0 - 128));
            } else {
                const float* p = rowA + (k0 - 256);
#pragma unroll
                for (int j = 0; j < 16; ++j) bfrag[j] = (_Float16)p[j];
            }
            const _Float16* fb = We1s + kt * 4096;
            v16h af[8];
#pragma unroll
            for (int t = 0; t < 8; ++t) af[t] = ldfrag(fb, t, lane);
#pragma unroll
            for (int t = 0; t < 8; ++t) acc[t] = wmma_f16(af[t], bfrag, acc[t]);
        }

        // bias + ReLU -> f16 scratch, row-major [m][h], pitch 144
#pragma unroll
        for (int t = 0; t < 8; ++t) {
            v8f bias = *(const v8f*)(be1 + t * 16 + lh * 8);
            v8f r = acc[t] + bias;
            v8h h8;
#pragma unroll
            for (int j = 0; j < 8; ++j) h8[j] = (_Float16)fmaxf(r[j], 0.0f);
            *(v8h*)(sc + m * 144 + t * 16 + lh * 8) = h8;
        }

        v8f acc2[8] = {};
#pragma unroll
        for (int kt = 0; kt < 4; ++kt) {
            v16h bfrag = *(const v16h*)(sc + m * 144 + kt * 32 + lh * 16);
            const _Float16* fb = We2t_g + kt * 4096;
            v16h af[8];
#pragma unroll
            for (int t = 0; t < 8; ++t) af[t] = ldfrag(fb, t, lane);
#pragma unroll
            for (int t = 0; t < 8; ++t) acc2[t] = wmma_f16(af[t], bfrag, acc2[t]);
        }

        float* eRow = eout + ((size_t)b * EE + eI) * HH;
        float* aRow = agg  + ((size_t)b * NN + di) * HH;
#pragma unroll
        for (int t = 0; t < 8; ++t) {
            v8f bias = *(const v8f*)(be2 + t * 16 + lh * 8);
            v8f r = acc2[t] + bias;
            *(v8f*)(eRow + t * 16 + lh * 8) = r;
#pragma unroll
            for (int j = 0; j < 8; ++j)
                unsafeAtomicAdd(aRow + t * 16 + lh * 8 + j, r[j]);
        }
    }
}

// ---------------- node MLP ----------------
__global__ __launch_bounds__(256, 4) void node_kernel(
    const _Float16* __restrict__ xh,
    const float* __restrict__ agg,
    const _Float16* __restrict__ Wn1t_g,    // frag-major, Wn2t follows contiguously
    const float* __restrict__ bn1,
    const float* __restrict__ bn2,
    float* __restrict__ xout)
{
    extern __shared__ _Float16 smem[];      // [Wn1t][Wn2t][scratch]
    const int wave = threadIdx.x >> 5;
    const int lane = threadIdx.x & 31;
    const int m    = lane & 15;
    const int lh   = lane >> 4;

    if (wave == 0) {
        tdm_load_1d(Wn1t_g, (unsigned)(unsigned long long)(void*)smem, 49152 * 2);
        __builtin_amdgcn_s_wait_tensorcnt(0);
    }
    __syncthreads();

    const int tile = blockIdx.x * 8 + wave;
    if (tile >= BB * (NN / 16)) return;
    const int b  = tile / (NN / 16);
    const int n0 = (tile % (NN / 16)) * 16;

    const _Float16* Wn1s = smem;
    const _Float16* Wn2s = smem + LDS_N_WN2;
    _Float16* sc = smem + LDS_N_SC + wave * (16 * 144);

    const _Float16* rowX = xh  + ((size_t)b * NN + n0 + m) * HH;
    const float*    rowG = agg + ((size_t)b * NN + n0 + m) * HH;

    v8f acc[8] = {};
#pragma unroll
    for (int kt = 0; kt < 8; ++kt) {
        const int k0 = kt * 32 + lh * 16;
        v16h bfrag;
        if (kt < 4) {
            bfrag = *(const v16h*)(rowX + k0);
        } else {
            const float* p = rowG + (k0 - 128);
#pragma unroll
            for (int j = 0; j < 16; ++j) bfrag[j] = (_Float16)p[j];
        }
        const _Float16* fb = Wn1s + kt * 4096;
        v16h af[8];
#pragma unroll
        for (int t = 0; t < 8; ++t) af[t] = ldfrag(fb, t, lane);
#pragma unroll
        for (int t = 0; t < 8; ++t) acc[t] = wmma_f16(af[t], bfrag, acc[t]);
    }

#pragma unroll
    for (int t = 0; t < 8; ++t) {
        v8f bias = *(const v8f*)(bn1 + t * 16 + lh * 8);
        v8f r = acc[t] + bias;
        v8h h8;
#pragma unroll
        for (int j = 0; j < 8; ++j) h8[j] = (_Float16)fmaxf(r[j], 0.0f);
        *(v8h*)(sc + m * 144 + t * 16 + lh * 8) = h8;
    }

    v8f acc2[8] = {};
#pragma unroll
    for (int kt = 0; kt < 4; ++kt) {
        v16h bfrag = *(const v16h*)(sc + m * 144 + kt * 32 + lh * 16);
        const _Float16* fb = Wn2s + kt * 4096;
        v16h af[8];
#pragma unroll
        for (int t = 0; t < 8; ++t) af[t] = ldfrag(fb, t, lane);
#pragma unroll
        for (int t = 0; t < 8; ++t) acc2[t] = wmma_f16(af[t], bfrag, acc2[t]);
    }

    float* oRow = xout + ((size_t)b * NN + n0 + m) * HH;
#pragma unroll
    for (int t = 0; t < 8; ++t) {
        v8f bias = *(const v8f*)(bn2 + t * 16 + lh * 8);
        v8f r = acc2[t] + bias;
        *(v8f*)(oRow + t * 16 + lh * 8) = r;
    }
}

extern "C" void kernel_launch(void* const* d_in, const int* in_sizes, int n_in,
                              void* d_out, int out_size, void* d_ws, size_t ws_size,
                              hipStream_t stream) {
    const float*     x    = (const float*)d_in[0];
    const long long* eidx = (const long long*)d_in[1];
    const float*     ea   = (const float*)d_in[2];
    const float*     We1  = (const float*)d_in[3];
    const float*     be1  = (const float*)d_in[4];
    const float*     We2  = (const float*)d_in[5];
    const float*     be2  = (const float*)d_in[6];
    const float*     Wn1  = (const float*)d_in[7];
    const float*     bn1  = (const float*)d_in[8];
    const float*     Wn2  = (const float*)d_in[9];
    const float*     bn2  = (const float*)d_in[10];

    char* ws = (char*)d_ws;
    _Float16* xhp = (_Float16*)(ws + WS_XH);
    float*    agg = (float*)(ws + WS_AGG);
    _Float16* wt  = (_Float16*)(ws + WS_WT);

    float* xout = (float*)d_out;
    float* eout = (float*)d_out + (size_t)BB * NN * HH;

    prep_weights_kernel<<<dim3(192, 4), 256, 0, stream>>>(We1, We2, Wn1, Wn2, wt);
    prep_x_kernel<<<6250, 256, 0, stream>>>(x, xhp);
    (void)hipMemsetAsync(agg, 0, (size_t)BB * NN * HH * sizeof(float), stream);

    const unsigned edgeLds = (LDS_E_SC + 8 * 16 * 144) * 2;          // 135168 B
    const unsigned nodeLds = (LDS_N_SC + 8 * 16 * 144) * 2;          // 135168 B
    edge_kernel<<<1250, 256, edgeLds, stream>>>(xhp, eidx, ea,
                                                wt, be1, wt + OFF_WE2T, be2,
                                                agg, eout);
    node_kernel<<<782, 256, nodeLds, stream>>>(xhp, agg,
                                               wt + OFF_WN1T, bn1, bn2,
                                               xout);
}
